// Ernie4_5_VLMoE_25933012533747
// MI455X (gfx1250) — matrix operand order repaired and checked by reading, not compiled, not run on hardware
//
#include <hip/hip_runtime.h>
#include <hip/hip_bf16.h>

#define T_TOK 2048
#define HDIM  2560
#define NEXP  8
#define I_TXT 1536
#define I_IMG 512
#define I_SH  3072

typedef __attribute__((ext_vector_type(16))) __bf16 v16bf;
typedef __attribute__((ext_vector_type(8)))  float  v8f;

union V16 {
  v16bf v;
  unsigned u[8];
  unsigned short s[16];
  uint4 q[2];
};

// Pack two fp32 into one dword of bf16 (round-half-up): 2x v_add + 1x v_perm_b32
__device__ __forceinline__ unsigned pack2bf(float a, float b) {
  unsigned ua = __float_as_uint(a) + 0x8000u;
  unsigned ub = __float_as_uint(b) + 0x8000u;
  return __builtin_amdgcn_perm(ub, ua, 0x07060302u);  // {ua[31:16], ub[31:16]}
}
__device__ __forceinline__ unsigned short f2bf1(float a) {
  return (unsigned short)((__float_as_uint(a) + 0x8000u) >> 16);
}

__device__ __forceinline__ v8f wmma_bf16(const V16& A, const V16& B, v8f c) {
  return __builtin_amdgcn_wmma_f32_16x16x32_bf16(false, A.v, false, B.v,
                                                 (short)0, c, false, false);
}

// ---------------------------------------------------------------------------
// One-shot activation conversion: x fp32 [T,H] -> bf16 copy in workspace.
// Removes all A-side conversion VALU from the GEMM inner loops.
// ---------------------------------------------------------------------------
__global__ __launch_bounds__(256)
void convert_x_kernel(const float* __restrict__ x, unsigned* __restrict__ xb) {
  const size_t idx = (size_t)blockIdx.x * 256 + threadIdx.x;   // dword index
  if (idx >= (size_t)T_TOK * HDIM / 2) return;
  const float2 v = *(const float2*)(x + 2 * idx);
  xb[idx] = pack2bf(v.x, v.y);
}

// ---------------------------------------------------------------------------
// Router: softmax over 8 logits, top-2, renormalize, fold in modality mask.
// ---------------------------------------------------------------------------
__device__ __forceinline__ void write_top2(const float* l, float* w, float scale) {
  float mx = l[0];
  #pragma unroll
  for (int e = 1; e < NEXP; ++e) mx = fmaxf(mx, l[e]);
  float p[NEXP];
  #pragma unroll
  for (int e = 0; e < NEXP; ++e) p[e] = __expf(l[e] - mx);
  int i1 = 0;
  #pragma unroll
  for (int e = 1; e < NEXP; ++e) if (p[e] > p[i1]) i1 = e;
  int i2 = (i1 == 0) ? 1 : 0;
  #pragma unroll
  for (int e = 0; e < NEXP; ++e) if (e != i1 && p[e] > p[i2]) i2 = e;
  float rden = __builtin_amdgcn_rcpf(p[i1] + p[i2]);
  #pragma unroll
  for (int e = 0; e < NEXP; ++e) {
    float v = (e == i1) ? p[i1] * rden : ((e == i2) ? p[i2] * rden : 0.0f);
    w[e] = v * scale;
  }
}

__global__ __launch_bounds__(256)
void router_kernel(const float* __restrict__ x,
                   const unsigned char* __restrict__ vmask,   // jax bool: 1 byte
                   const float* __restrict__ tgate,           // [8,H]
                   const float* __restrict__ igate,           // [8,H]
                   float* __restrict__ wt,                    // [T,8]
                   float* __restrict__ wi)                    // [T,8]
{
  const int lane = threadIdx.x & 31;
  const int t    = blockIdx.x * 8 + (threadIdx.x >> 5);
  if (t >= T_TOK) return;

  float at[NEXP] = {0,0,0,0,0,0,0,0};
  float ai[NEXP] = {0,0,0,0,0,0,0,0};
  for (int h = lane; h < HDIM; h += 32) {
    float xv = x[(size_t)t * HDIM + h];
    #pragma unroll
    for (int e = 0; e < NEXP; ++e) {
      at[e] = fmaf(xv, tgate[e * HDIM + h], at[e]);
      ai[e] = fmaf(xv, igate[e * HDIM + h], ai[e]);
    }
  }
  #pragma unroll
  for (int e = 0; e < NEXP; ++e) {
    #pragma unroll
    for (int s = 16; s > 0; s >>= 1) {
      at[e] += __shfl_xor(at[e], s, 32);
      ai[e] += __shfl_xor(ai[e], s, 32);
    }
  }
  if (lane == 0) {
    const bool vis = vmask[t] != 0;
    write_top2(at, wt + (size_t)t * NEXP, vis ? 0.0f : 1.0f);
    write_top2(ai, wi + (size_t)t * NEXP, vis ? 1.0f : 0.0f);
  }
}

// ---------------------------------------------------------------------------
// Fused SwiGLU expert tile over MT*16 tokens (MT M-tiles), 8 waves / block.
// Phase 1: h[MT*16,I] = silu(x Wg)*(x Wu) into LDS (bf16). A operands are
// direct bf16 loads from the pre-converted x copy; each B tile (converted
// fp32->bf16 in registers) feeds MT WMMAs.
// Phase 2: out[MT*16,H] (+)= diag(w) * (h Wd); gated -> atomic add, else store.
// LDS: [0..255] w_s[MT*16], [256..319] skip flag, [320..) h bf16 [MT*16, I].
// ---------------------------------------------------------------------------
template <int MT>
__global__ __launch_bounds__(256)
void moe_tile_kernel(const unsigned short* __restrict__ xb,  // bf16 x [T,H]
                     const float* __restrict__ gatew,        // [T,8] or null
                     const float* __restrict__ wg_all,       // [E,H,I]
                     const float* __restrict__ wu_all,
                     const float* __restrict__ wd_all,       // [E,I,H]
                     float* __restrict__ out,
                     int I, int gated)
{
  extern __shared__ char smem[];
  float*          w_s    = (float*)smem;             // MT*16 gate weights
  int*            skip_s = (int*)(smem + 256);
  unsigned short* hs     = (unsigned short*)(smem + 320);

  const int tid  = threadIdx.x;
  const int lane = tid & 31;
  const int wv   = tid >> 5;          // 0..7
  const int t0   = blockIdx.x * (MT * 16);
  const int e    = blockIdx.y;

  const float* wg = wg_all + (size_t)e * HDIM * I;
  const float* wu = wu_all + (size_t)e * HDIM * I;
  const float* wd = wd_all + (size_t)e * I * HDIM;

  if (tid == 0) skip_s[0] = 0;
  __syncthreads();
  if (tid < MT * 16) {
    float w = gated ? gatew[(size_t)(t0 + tid) * NEXP + e] : 1.0f;
    w_s[tid] = w;
    if (w != 0.0f) atomicOr(skip_s, 1);
  }
  __syncthreads();
  if (gated && skip_s[0] == 0) return;   // whole token tile skips expert e

  const int m    = lane & 15;   // A row within M tile == B/C column within N tile
  const int half = lane >> 4;   // K-half per ISA 16-bit operand layouts

  // ---- phase 1: h = silu(x Wg) * (x Wu) -----------------------------------
  for (int nt = wv; nt < (I >> 4); nt += 8) {
    const int n = (nt << 4) + m;
    v8f accg[MT], accu[MT];
    #pragma unroll
    for (int t = 0; t < MT; ++t) { accg[t] = (v8f){}; accu[t] = (v8f){}; }

    for (int k0 = 0; k0 < HDIM; k0 += 32) {
      V16 A[MT], Bg, Bu;
      #pragma unroll
      for (int t = 0; t < MT; ++t) {   // direct bf16 loads, no conversion
        const unsigned short* ar =
            xb + (size_t)(t0 + t * 16 + m) * HDIM + k0 + half * 8;
        A[t].q[0] = *(const uint4*)(ar);
        A[t].q[1] = *(const uint4*)(ar + 16);
      }
      __builtin_prefetch(wg + (size_t)(k0 + 32) * I + n, 0, 0);
      __builtin_prefetch(wu + (size_t)(k0 + 32) * I + n, 0, 0);
      #pragma unroll
      for (int r = 0; r < 8; ++r) {
        const int ka = k0 + half * 8 + 2 * (r & 3) + (r >> 2) * 16;
        Bg.u[r] = pack2bf(wg[(size_t)ka * I + n], wg[(size_t)(ka + 1) * I + n]);
        Bu.u[r] = pack2bf(wu[(size_t)ka * I + n], wu[(size_t)(ka + 1) * I + n]);
      }
      #pragma unroll
      for (int t = 0; t < MT; ++t) {
        accg[t] = wmma_bf16(A[t], Bg, accg[t]);
        accu[t] = wmma_bf16(A[t], Bu, accu[t]);
      }
    }
    // C layout: VGPR i -> M = i + half*8, N = lane&15
    #pragma unroll
    for (int t = 0; t < MT; ++t) {
      #pragma unroll
      for (int i = 0; i < 8; ++i) {
        const int row = i + half * 8;
        float g = accg[t][i];
        float h = g * __builtin_amdgcn_rcpf(1.0f + __expf(-g)) * accu[t][i];
        hs[(size_t)(t * 16 + row) * I + n] = f2bf1(h);
      }
    }
  }
  __syncthreads();

  // ---- phase 2: out (+)= diag(w) * (h Wd) ---------------------------------
  for (int nt = wv; nt < (HDIM >> 4); nt += 8) {
    const int n = (nt << 4) + m;
    v8f c[MT];
    #pragma unroll
    for (int t = 0; t < MT; ++t) c[t] = (v8f){};

    for (int k0 = 0; k0 < I; k0 += 32) {
      V16 A[MT], B;
      #pragma unroll
      for (int t = 0; t < MT; ++t) {
        const unsigned short* ar = hs + (size_t)(t * 16 + m) * I + k0 + half * 8;
        A[t].q[0] = *(const uint4*)(ar);
        A[t].q[1] = *(const uint4*)(ar + 16);
      }
      __builtin_prefetch(wd + (size_t)(k0 + 32) * HDIM + n, 0, 0);
      #pragma unroll
      for (int r = 0; r < 8; ++r) {
        const int ka = k0 + half * 8 + 2 * (r & 3) + (r >> 2) * 16;
        B.u[r] = pack2bf(wd[(size_t)ka * HDIM + n], wd[(size_t)(ka + 1) * HDIM + n]);
      }
      #pragma unroll
      for (int t = 0; t < MT; ++t) c[t] = wmma_bf16(A[t], B, c[t]);
    }
    #pragma unroll
    for (int t = 0; t < MT; ++t) {
      #pragma unroll
      for (int i = 0; i < 8; ++i) {
        const int row = t * 16 + i + half * 8;
        float v = c[t][i] * w_s[row];
        float* dst = out + (size_t)(t0 + row) * HDIM + n;
        if (gated) unsafeAtomicAdd(dst, v);   // global_atomic_add_f32
        else       *dst = v;                  // shared expert initializes d_out
      }
    }
  }
}

// ---------------------------------------------------------------------------
extern "C" void kernel_launch(void* const* d_in, const int* in_sizes, int n_in,
                              void* d_out, int out_size, void* d_ws, size_t ws_size,
                              hipStream_t stream) {
  const float*         x     = (const float*)d_in[0];
  const unsigned char* vmask = (const unsigned char*)d_in[1];
  const float*         tgw   = (const float*)d_in[2];
  const float*         twg   = (const float*)d_in[3];
  const float*         twu   = (const float*)d_in[4];
  const float*         twd   = (const float*)d_in[5];
  const float*         igw   = (const float*)d_in[6];
  const float*         iwg   = (const float*)d_in[7];
  const float*         iwu   = (const float*)d_in[8];
  const float*         iwd   = (const float*)d_in[9];
  const float*         swg   = (const float*)d_in[10];
  const float*         swu   = (const float*)d_in[11];
  const float*         swd   = (const float*)d_in[12];
  float* out = (float*)d_out;

  // workspace layout: gate weights (fp32) + bf16 copy of x (~10.6 MB total)
  float*          wt = (float*)d_ws;                    // [T,8] text (masked)
  float*          wi = wt + (size_t)T_TOK * NEXP;       // [T,8] image (masked)
  unsigned*       xb = (unsigned*)(wi + (size_t)T_TOK * NEXP);  // bf16 [T,H]
  const unsigned short* xbs = (const unsigned short*)xb;

  // 1) convert activations to bf16 once (A-side of every GEMM)
  convert_x_kernel<<<(T_TOK * HDIM / 2 + 255) / 256, 256, 0, stream>>>(x, xb);

  // 2) route both modalities (visual mask folded into gate weights)
  router_kernel<<<T_TOK / 8, 256, 0, stream>>>(x, vmask, tgw, igw, wt, wi);

  // 3) shared experts: dense SwiGLU, plain stores -> initializes d_out
  //    MT=2 (32 tokens): LDS h = 32*3072*2 = 192 KB
  moe_tile_kernel<2><<<dim3(T_TOK / 32, 1), 256, 320 + (size_t)32 * I_SH * 2, stream>>>(
      xbs, nullptr, swg, swu, swd, out, I_SH, 0);

  // 4) text experts: MT=4 (64 tokens): LDS h = 64*1536*2 = 192 KB
  moe_tile_kernel<4><<<dim3(T_TOK / 64, NEXP), 256, 320 + (size_t)64 * I_TXT * 2, stream>>>(
      xbs, wt, twg, twu, twd, out, I_TXT, 1);

  // 5) image experts: MT=4 (64 tokens): LDS h = 64*512*2 = 64 KB
  moe_tile_kernel<4><<<dim3(T_TOK / 64, NEXP), 256, 320 + (size_t)64 * I_IMG * 2, stream>>>(
      xbs, wi, iwg, iwu, iwd, out, I_IMG, 1);
}
